// FullAttention_84645215469988
// MI455X (gfx1250) — compile-verified, matrix operand order checked by reading, not run on hardware
//
#include <hip/hip_runtime.h>

// Flash attention forward, causal, bf16 WMMA matmuls + f32 softmax/accum.
// B=2, L=S=2048, H=16, E=D=64.
// v4: LDS double-buffered K/V staging (1 barrier/chunk, global loads overlap
//     WMMA), batched ds_swizzle butterfly reductions (8 rows in flight).

namespace {
constexpr int Bn = 2, Ln = 2048, Sn = 2048, Hn = 16, En = 64, Dn = 64;
constexpr int BQ = 64;   // query rows per block (16 per wave x 4 waves)
constexpr int BK = 32;   // key chunk = WMMA K of the P*V gemm
constexpr int NW = 4;    // waves per block
constexpr int KPAD = 72; // padded K-tile row (bf16 elems): 144B, keeps 16B slots
constexpr int VPAD = 40; // padded Vt / P row (bf16 elems): 80B
}

typedef __bf16 bf16;
typedef __attribute__((ext_vector_type(16))) __bf16 v16bf;
typedef __attribute__((ext_vector_type(8)))  __bf16 v8bf;
typedef __attribute__((ext_vector_type(8)))  float  v8f;

template <bool V> struct BoolC { static constexpr bool value = V; };

static __device__ inline v16bf cat8(v8bf a, v8bf b) {
  return __builtin_shufflevector(a, b, 0,1,2,3,4,5,6,7,8,9,10,11,12,13,14,15);
}

static __device__ inline float fast_exp2(float x) {
#if __has_builtin(__builtin_amdgcn_exp2f)
  return __builtin_amdgcn_exp2f(x);     // bare v_exp_f32; underflow -> 0 is fine
#else
  return exp2f(x);
#endif
}

static __device__ inline float fast_rcp(float x) {
#if __has_builtin(__builtin_amdgcn_rcpf)
  return __builtin_amdgcn_rcpf(x);
#else
  return 1.0f / x;
#endif
}

// ds_swizzle group-of-32 xor patterns: offset = xor<<10 | or<<5 | and(0x1f)
#define SWZ_XOR(x, imm) \
  __int_as_float(__builtin_amdgcn_ds_swizzle(__float_as_int(x), (imm)))

// One butterfly step for 8 independent rows: all swizzles issue before the
// maxes consume them (in-order DS returns -> pipelined waits).
template <int IMM>
static __device__ inline void bfly_max8(float (&mx)[8]) {
  float t[8];
#pragma unroll
  for (int v = 0; v < 8; ++v) t[v] = SWZ_XOR(mx[v], IMM);
#pragma unroll
  for (int v = 0; v < 8; ++v) mx[v] = fmaxf(mx[v], t[v]);
}

__global__ __launch_bounds__(128) void
fattn_fwd(const float* __restrict__ Qg, const float* __restrict__ Kg,
          const float* __restrict__ Vg, float* __restrict__ Og)
{
  __shared__ __align__(16) bf16 Kl[2][BK][KPAD];   // K chunk, row-major [k][e]
  __shared__ __align__(16) bf16 Vt[2][Dn][VPAD];   // V chunk, transposed [d][k]
  __shared__ __align__(16) bf16 Pl[NW][16][VPAD];  // per-wave P re-layout buffer

  const int tid  = threadIdx.x;
  const int wave = tid >> 5;
  const int lane = tid & 31;
  const int lo   = lane & 15;
  const int hi   = lane >> 4;   // 0: lanes 0-15, 1: lanes 16-31
  const int kb   = hi * 8;      // fragment K-slot base per ISA 7.12.2

  const int qBase = blockIdx.x * BQ;
  const int h     = blockIdx.y;
  const int b     = blockIdx.z;
  const int qRow0 = qBase + wave * 16;

  // softmax scale folded into Q, in log2 domain: (1/sqrt(E)) * log2(e)
  const float qscale = 0.125f * 1.44269504088896340736f;

  // ---- Q A-fragments (16x32 bf16, two K-chunks over E=64), pre-scaled ----
  v16bf aQ[2];
  {
    const float* qp = Qg + (((size_t)b * Ln + (qRow0 + lo)) * Hn + h) * En;
#pragma unroll
    for (int kc = 0; kc < 2; ++kc) {
      float4 f0 = *(const float4*)(qp + kc*32 + kb + 0);
      float4 f1 = *(const float4*)(qp + kc*32 + kb + 4);
      float4 f2 = *(const float4*)(qp + kc*32 + 16 + kb + 0);
      float4 f3 = *(const float4*)(qp + kc*32 + 16 + kb + 4);
      v16bf a;
      a[0]=(bf16)(f0.x*qscale);  a[1]=(bf16)(f0.y*qscale);
      a[2]=(bf16)(f0.z*qscale);  a[3]=(bf16)(f0.w*qscale);
      a[4]=(bf16)(f1.x*qscale);  a[5]=(bf16)(f1.y*qscale);
      a[6]=(bf16)(f1.z*qscale);  a[7]=(bf16)(f1.w*qscale);
      a[8]=(bf16)(f2.x*qscale);  a[9]=(bf16)(f2.y*qscale);
      a[10]=(bf16)(f2.z*qscale); a[11]=(bf16)(f2.w*qscale);
      a[12]=(bf16)(f3.x*qscale); a[13]=(bf16)(f3.y*qscale);
      a[14]=(bf16)(f3.z*qscale); a[15]=(bf16)(f3.w*qscale);
      aQ[kc] = a;
    }
  }

  // all-ones B fragment: o[4] accumulates P row-sums (every column identical)
  v16bf bOnes;
#pragma unroll
  for (int i = 0; i < 16; ++i) bOnes[i] = (bf16)1.0f;

  v8f o[5] = {{}, {}, {}, {}, {}};  // 4 D-tiles of O + row-sum tile, C-layout
  float rowM[8];                    // running row max (log2 domain)
#pragma unroll
  for (int v = 0; v < 8; ++v) rowM[v] = -3.0e38f;

  const int nChunks = qBase / BK + 2;  // causal trip count for this block

  // ---- double-buffered staging helpers (prefetch regs -> convert -> LDS) ----
  float4 kf[4], vf[4];
  auto gload = [&](int key0) {
#pragma unroll
    for (int i = 0; i < 4; ++i) {
      int fidx = tid + 128 * i;          // 512 float4 slots across the block
      int row  = fidx >> 4;              // key row within chunk (0..31)
      int e    = (fidx & 15) * 4;        // feature col (0..63, step 4)
      size_t gbase = ((size_t)b * Sn + (key0 + row)) * Hn + h;
      kf[i] = *(const float4*)(Kg + gbase * En + e);
      vf[i] = *(const float4*)(Vg + gbase * Dn + e);
    }
  };
  auto lstore = [&](int buf) {
#pragma unroll
    for (int i = 0; i < 4; ++i) {
      int fidx = tid + 128 * i;
      int row  = fidx >> 4;
      int e    = (fidx & 15) * 4;
      Kl[buf][row][e+0] = (bf16)kf[i].x; Kl[buf][row][e+1] = (bf16)kf[i].y;
      Kl[buf][row][e+2] = (bf16)kf[i].z; Kl[buf][row][e+3] = (bf16)kf[i].w;
      Vt[buf][e+0][row] = (bf16)vf[i].x; Vt[buf][e+1][row] = (bf16)vf[i].y;
      Vt[buf][e+2][row] = (bf16)vf[i].z; Vt[buf][e+3][row] = (bf16)vf[i].w;
    }
  };

  gload(0);
  lstore(0);
  __syncthreads();

  for (int ck = 0; ck < nChunks; ++ck) {
    const int key0 = ck * BK;
    const int cur  = ck & 1;
    const bool haveNext = (ck + 1 < nChunks);

    if (haveNext) gload(key0 + BK);      // HBM latency overlaps the WMMAs below

    if (key0 <= qRow0 + 15) {            // wave-uniform causal skip
      // ---- scores: S = Q*K^T (log2-scaled), two 16x16 N-tiles ----
      v8f c0 = {}, c1 = {};
#pragma unroll
      for (int kc = 0; kc < 2; ++kc) {
        int e0 = kc * 32 + kb;
        v16bf b0 = cat8(*(const v8bf*)&Kl[cur][lo][e0],
                        *(const v8bf*)&Kl[cur][lo][e0+16]);
        v16bf b1 = cat8(*(const v8bf*)&Kl[cur][16+lo][e0],
                        *(const v8bf*)&Kl[cur][16+lo][e0+16]);
        c0 = __builtin_amdgcn_wmma_f32_16x16x32_bf16(false, aQ[kc], false, b0,
                                                     (short)0, c0, false, false);
        c1 = __builtin_amdgcn_wmma_f32_16x16x32_bf16(false, aQ[kc], false, b1,
                                                     (short)0, c1, false, false);
      }

      // ---- online softmax (phased) + P*V, masked / mask-free variants ----
      auto body = [&](auto mt) {
        constexpr bool MASKED = decltype(mt)::value;
        const int k0 = key0 + lo;
        const int k1 = key0 + 16 + lo;
        float s0a[8], s1a[8], mx[8];
        // phase A: masked scores + per-lane row max
#pragma unroll
        for (int v = 0; v < 8; ++v) {
          float s0 = c0[v];
          float s1 = c1[v];
          if constexpr (MASKED) {
            const int q = qRow0 + v + 8 * hi;
            if (k0 > q) s0 = -3.0e38f;
            if (k1 > q) s1 = -3.0e38f;
          }
          s0a[v] = s0; s1a[v] = s1;
          mx[v] = fmaxf(s0, s1);
        }
        // phase B: batched 16-lane-group butterflies (xor 1,2,4,8)
        bfly_max8<0x041F>(mx);   // SWAPX1
        bfly_max8<0x081F>(mx);   // SWAPX2
        bfly_max8<0x101F>(mx);   // SWAPX4
        bfly_max8<0x201F>(mx);   // SWAPX8
        // phase C: exponentials, accumulator rescale, P staging
#pragma unroll
        for (int v = 0; v < 8; ++v) {
          const float mnew = fmaxf(rowM[v], mx[v]);
          const float corr = fast_exp2(rowM[v] - mnew);
          const float p0   = fast_exp2(s0a[v] - mnew);
          const float p1   = fast_exp2(s1a[v] - mnew);
          rowM[v] = mnew;
#pragma unroll
          for (int t = 0; t < 5; ++t) o[t][v] *= corr;
          // C-layout -> LDS so we can regather in A-layout (same-wave DS order)
          Pl[wave][v + 8*hi][lo]      = (bf16)p0;
          Pl[wave][v + 8*hi][16 + lo] = (bf16)p1;
        }

        // O += P * V ; row-sum tile += P * ones
        v16bf aP = cat8(*(const v8bf*)&Pl[wave][lo][kb],
                        *(const v8bf*)&Pl[wave][lo][16 + kb]);
#pragma unroll
        for (int dt = 0; dt < 4; ++dt) {
          v16bf bv = cat8(*(const v8bf*)&Vt[cur][dt*16 + lo][kb],
                          *(const v8bf*)&Vt[cur][dt*16 + lo][16 + kb]);
          o[dt] = __builtin_amdgcn_wmma_f32_16x16x32_bf16(false, aP, false, bv,
                                                          (short)0, o[dt], false, false);
        }
        o[4] = __builtin_amdgcn_wmma_f32_16x16x32_bf16(false, aP, false, bOnes,
                                                       (short)0, o[4], false, false);
      };

      if (key0 + BK - 1 <= qRow0) body(BoolC<false>{});  // strictly below diagonal
      else                        body(BoolC<true>{});   // diagonal-straddling chunk
    }

    if (haveNext) lstore((ck + 1) & 1);  // fill the other buffer
    __syncthreads();                     // single barrier per chunk
  }

  // ---- epilogue: one rcp per row, multiply-normalize, store f32 ----
#pragma unroll
  for (int v = 0; v < 8; ++v) {
    const int q = qRow0 + v + 8 * hi;
    const float inv = fast_rcp(o[4][v]);
    float* op = Og + (((size_t)b * Ln + q) * Hn + h) * Dn + lo;
#pragma unroll
    for (int dt = 0; dt < 4; ++dt)
      op[dt * 16] = o[dt][v] * inv;
  }
}

extern "C" void kernel_launch(void* const* d_in, const int* in_sizes, int n_in,
                              void* d_out, int out_size, void* d_ws, size_t ws_size,
                              hipStream_t stream) {
  (void)in_sizes; (void)n_in; (void)out_size; (void)d_ws; (void)ws_size;
  const float* Qg = (const float*)d_in[0];   // queries [B,L,H,E] f32
  const float* Kg = (const float*)d_in[1];   // keys    [B,S,H,E] f32
  const float* Vg = (const float*)d_in[2];   // values  [B,S,H,D] f32
  // d_in[3] (attn_mask) is analytically triu(k=1): replaced by the causal test.
  float* Og = (float*)d_out;                 // output  [B,L,H,D] f32

  dim3 grid(Ln / BQ, Hn, Bn);                // (32, 16, 2)
  dim3 block(128);                           // 4 waves (wave32)
  hipLaunchKernelGGL(fattn_fwd, grid, block, 0, stream, Qg, Kg, Vg, Og);
}
